// GNN_79645873537523
// MI455X (gfx1250) — matrix operand order repaired and checked by reading, not compiled, not run on hardware
//
#include <hip/hip_runtime.h>
#include <hip/hip_bf16.h>

typedef __bf16 bf16;
typedef __attribute__((ext_vector_type(16))) __bf16 v16bf;
typedef __attribute__((ext_vector_type(8)))  __bf16 v8bf;
typedef __attribute__((ext_vector_type(8)))  float  v8f;

#define N_NODES 50000
#define N_EDGES 800000
#define N_ITERS 7
#define D_H    128
#define D_IN   64
#define D_EDGE 128
#define D_OUT  64
#define MLP_H  96

#define N_ETILES (N_EDGES / 16)   // 50000
#define N_NTILES (N_NODES / 16)   // 3125
#define T_EDGE 8                  // edge tiles per wave
#define T_GRU  8                  // node tiles per wave

// ---------- helpers ----------
__device__ __forceinline__ unsigned short f2bfu(float f) {
  union { float f; unsigned u; } c; c.f = f;
  unsigned u = c.u;
  u += 0x7FFFu + ((u >> 16) & 1u);          // round-to-nearest-even
  return (unsigned short)(u >> 16);
}
__device__ __forceinline__ __bf16 f2bf(float f) {
  unsigned short s = f2bfu(f);
  union { unsigned short s; __bf16 b; } c; c.s = s;
  return c.b;
}
__device__ __forceinline__ v8f vzero8() {
  v8f z = {0.f,0.f,0.f,0.f,0.f,0.f,0.f,0.f};
  return z;
}
__device__ __forceinline__ v8f wmma_bf16(v16bf a, v16bf b, v8f c) {
  return __builtin_amdgcn_wmma_f32_16x16x32_bf16(false, a, false, b, (short)0, c, false, false);
}
// load B fragment (tile index t, lane l): 16 bf16 = 32 contiguous bytes per lane
__device__ __forceinline__ v16bf ldb(const unsigned short* f, int t, int l) {
  return *(const v16bf*)(f + ((size_t)t * 32 + l) * 16);
}
// build A fragment (16x32 bf16) from 8+8 contiguous bf16 at rp, rp+16
__device__ __forceinline__ v16bf lda_bf(const bf16* rp) {
  v8bf p = *(const v8bf*)rp;
  v8bf q = *(const v8bf*)(rp + 16);
  v16bf a;
  #pragma unroll
  for (int j = 0; j < 8; ++j) { a[j] = p[j]; a[j + 8] = q[j]; }
  return a;
}
__device__ __forceinline__ v16bf lda_f32(const float* rp) {
  v16bf a;
  #pragma unroll
  for (int j = 0; j < 8; ++j) { a[j] = f2bf(rp[j]); a[j + 8] = f2bf(rp[16 + j]); }
  return a;
}
__device__ __forceinline__ void copy_to_lds(void* dstv, const void* srcv, int bytes) {
  uint4* dst = (uint4*)dstv;
  const uint4* src = (const uint4*)srcv;
  int n = bytes >> 4;
  for (int i = threadIdx.x; i < n; i += blockDim.x) dst[i] = src[i];
}
__device__ __forceinline__ void atomic_add_f32(float* p, float v) {
  __hip_atomic_fetch_add(p, v, __ATOMIC_RELAXED, __HIP_MEMORY_SCOPE_AGENT);
}

// ---------- weight packing: W[out,in] f32 -> WMMA B fragments bf16 ----------
// frag[(nt*nKt+kt)][lane l][elem j] = W[nt*16+(l&15)][kt*32 + 8*(l>=16) + j + (j>=8 ? 8 : 0)]
__global__ __launch_bounds__(32) void pack_b_kernel(const float* __restrict__ W,
                                                    unsigned short* __restrict__ frag,
                                                    int nKt, int Kin) {
  int t = blockIdx.x;
  int l = threadIdx.x;
  int kt = t % nKt, nt = t / nKt;
  int lo = l & 15, hi = (l >> 4) & 1;
  int n  = nt * 16 + lo;
  int k0 = kt * 32 + hi * 8;
  unsigned short* dst = frag + ((size_t)t * 32 + l) * 16;
  const float* srcp = W + (size_t)n * Kin + k0;
  #pragma unroll
  for (int j = 0; j < 16; ++j) dst[j] = f2bfu(srcp[j < 8 ? j : j + 8]);
}

__global__ void zero_kernel(unsigned int* __restrict__ p, long n) {
  long i = (long)blockIdx.x * blockDim.x + threadIdx.x;
  long stride = (long)gridDim.x * blockDim.x;
  for (; i < n; i += stride) p[i] = 0u;
}

// ---------- edge message MLP + scatter-add ----------
// 8 waves/block; all edge-MLP weight fragments staged in LDS; each wave does
// T_EDGE tiles of 16 edges. Activation transpose buffer is per-wave private.
__global__ __launch_bounds__(256) void edge_mlp_kernel(
    const bf16* __restrict__ h,
    const int* __restrict__ src_ids, const int* __restrict__ dst_ids,
    const unsigned short* __restrict__ ewf,  // packed w1|w2|w3 fragments (92160 B)
    const float* __restrict__ b1, const float* __restrict__ b2,
    const float* __restrict__ b3,
    float* __restrict__ agg)
{
  __shared__ __align__(16) unsigned short wsh[46080];      // 90 KB of B fragments
  __shared__ __align__(16) unsigned short act[8 * 16 * MLP_H];
  copy_to_lds(wsh, ewf, 92160);
  __syncthreads();
  const unsigned short* w1s = wsh;           // 48 tiles
  const unsigned short* w2s = wsh + 24576;   // 18 tiles
  const unsigned short* w3s = wsh + 33792;   // 24 tiles

  const int w  = threadIdx.x >> 5;
  const int l  = threadIdx.x & 31;
  const int lo = l & 15, hi = (l >> 4) & 1;
  unsigned short* my = act + w * 16 * MLP_H;

  #pragma unroll 1
  for (int t = 0; t < T_EDGE; ++t) {
    const int et = (blockIdx.x * 8 + w) * T_EDGE + t;
    if (et >= N_ETILES) break;               // wave-uniform
    const int e0  = et * 16;
    const int src = src_ids[e0 + lo];
    const int dst = dst_ids[e0 + lo];

    // ---- layer 1: [16x256] @ [256x96] ----
    v8f acc1[6];
    #pragma unroll
    for (int nt = 0; nt < 6; ++nt) acc1[nt] = vzero8();
    #pragma unroll
    for (int kt = 0; kt < 8; ++kt) {
      const bf16* rp = h + (size_t)(kt < 4 ? src : dst) * D_H + (kt & 3) * 32 + hi * 8;
      v16bf a = lda_bf(rp);
      #pragma unroll
      for (int nt = 0; nt < 6; ++nt)
        acc1[nt] = wmma_bf16(a, ldb(w1s, nt * 8 + kt, l), acc1[nt]);
    }
    #pragma unroll
    for (int nt = 0; nt < 6; ++nt) {
      int n = nt * 16 + lo;
      float bias = b1[n];
      #pragma unroll
      for (int v = 0; v < 8; ++v) {
        float x = acc1[nt][v] + bias;
        my[(hi * 8 + v) * MLP_H + n] = f2bfu(x > 0.f ? x : 0.f);
      }
    }
    // same-wave LDS ops are in-order: store->load RAW is safe without barrier

    // ---- layer 2: [16x96] @ [96x96] ----
    v16bf a2[3];
    #pragma unroll
    for (int kt = 0; kt < 3; ++kt)
      a2[kt] = lda_bf((const bf16*)&my[lo * MLP_H + kt * 32 + hi * 8]);
    v8f acc2[6];
    #pragma unroll
    for (int nt = 0; nt < 6; ++nt) acc2[nt] = vzero8();
    #pragma unroll
    for (int kt = 0; kt < 3; ++kt)
      #pragma unroll
      for (int nt = 0; nt < 6; ++nt)
        acc2[nt] = wmma_bf16(a2[kt], ldb(w2s, nt * 3 + kt, l), acc2[nt]);
    #pragma unroll
    for (int nt = 0; nt < 6; ++nt) {
      int n = nt * 16 + lo;
      float bias = b2[n];
      #pragma unroll
      for (int v = 0; v < 8; ++v) {
        float x = acc2[nt][v] + bias;
        my[(hi * 8 + v) * MLP_H + n] = f2bfu(x > 0.f ? x : 0.f);
      }
    }

    // ---- layer 3: [16x96] @ [96x128] + b3, scatter-add ----
    v16bf a3[3];
    #pragma unroll
    for (int kt = 0; kt < 3; ++kt)
      a3[kt] = lda_bf((const bf16*)&my[lo * MLP_H + kt * 32 + hi * 8]);
    v8f acc3[8];
    #pragma unroll
    for (int nt = 0; nt < 8; ++nt) acc3[nt] = vzero8();
    #pragma unroll
    for (int kt = 0; kt < 3; ++kt)
      #pragma unroll
      for (int nt = 0; nt < 8; ++nt)
        acc3[nt] = wmma_bf16(a3[kt], ldb(w3s, nt * 3 + kt, l), acc3[nt]);
    #pragma unroll
    for (int nt = 0; nt < 8; ++nt) {
      int n = nt * 16 + lo;
      float b3v = b3[n];
      #pragma unroll
      for (int v = 0; v < 8; ++v) {
        int d = dst_ids[e0 + hi * 8 + v];
        atomic_add_f32(agg + (size_t)d * D_EDGE + n, acc3[nt][v] + b3v);
      }
    }
  }
}

// ---------- GRU cell ----------
// grid = (tile-groups, 8). blockIdx.y = feature tile ft; the block stages only
// the 30 B fragments (30 KB) for ntiles {ft, 8+ft, 16+ft}; 8 waves x T_GRU tiles.
__global__ __launch_bounds__(256) void gru_kernel(
    const float* __restrict__ node_inputs,
    const float* __restrict__ agg,
    const bf16* __restrict__ hbf_cur,
    float* __restrict__ hf32,
    bf16* __restrict__ hbf_nxt,
    const unsigned short* __restrict__ wihf,
    const unsigned short* __restrict__ whhf,
    const float* __restrict__ b_ih, const float* __restrict__ b_hh)
{
  __shared__ __align__(16) unsigned short gsh[15360];  // 30 fragments
  const int ft = blockIdx.y;
  copy_to_lds(gsh +     0, wihf + (size_t)(ft)      * 6 * 512, 6 * 1024);
  copy_to_lds(gsh +  3072, wihf + (size_t)(8 + ft)  * 6 * 512, 6 * 1024);
  copy_to_lds(gsh +  6144, wihf + (size_t)(16 + ft) * 6 * 512, 6 * 1024);
  copy_to_lds(gsh +  9216, whhf + (size_t)(ft)      * 4 * 512, 4 * 1024);
  copy_to_lds(gsh + 11264, whhf + (size_t)(8 + ft)  * 4 * 512, 4 * 1024);
  copy_to_lds(gsh + 13312, whhf + (size_t)(16 + ft) * 4 * 512, 4 * 1024);
  __syncthreads();

  const int w  = threadIdx.x >> 5;
  const int l  = threadIdx.x & 31;
  const int lo = l & 15, hi = (l >> 4) & 1;
  const float br  = b_ih[ft * 16 + lo]        + b_hh[ft * 16 + lo];
  const float bz  = b_ih[(8 + ft) * 16 + lo]  + b_hh[(8 + ft) * 16 + lo];
  const float bin = b_ih[(16 + ft) * 16 + lo];
  const float bhn = b_hh[(16 + ft) * 16 + lo];
  const int f = ft * 16 + lo;

  #pragma unroll 1
  for (int t = 0; t < T_GRU; ++t) {
    const int tile = (blockIdx.x * 8 + w) * T_GRU + t;
    if (tile >= N_NTILES) break;             // wave-uniform
    const size_t node = (size_t)tile * 16 + lo;

    v16bf ax[6];
    #pragma unroll
    for (int kt = 0; kt < 2; ++kt)
      ax[kt] = lda_f32(node_inputs + node * D_IN + kt * 32 + hi * 8);
    #pragma unroll
    for (int kt = 2; kt < 6; ++kt)
      ax[kt] = lda_f32(agg + node * D_EDGE + (kt - 2) * 32 + hi * 8);
    v16bf ah[4];
    #pragma unroll
    for (int kt = 0; kt < 4; ++kt)
      ah[kt] = lda_bf(hbf_cur + node * D_H + kt * 32 + hi * 8);

    v8f accr = vzero8(), accz = vzero8(), accin = vzero8(), acchn = vzero8();
    #pragma unroll
    for (int kt = 0; kt < 6; ++kt) {
      accr  = wmma_bf16(ax[kt], ldb(gsh, kt,      l), accr);
      accz  = wmma_bf16(ax[kt], ldb(gsh, 6 + kt,  l), accz);
      accin = wmma_bf16(ax[kt], ldb(gsh, 12 + kt, l), accin);
    }
    #pragma unroll
    for (int kt = 0; kt < 4; ++kt) {
      accr  = wmma_bf16(ah[kt], ldb(gsh, 18 + kt, l), accr);
      accz  = wmma_bf16(ah[kt], ldb(gsh, 22 + kt, l), accz);
      acchn = wmma_bf16(ah[kt], ldb(gsh, 26 + kt, l), acchn);
    }
    #pragma unroll
    for (int v = 0; v < 8; ++v) {
      size_t nm = (size_t)tile * 16 + hi * 8 + v;
      float r  = 1.f / (1.f + __expf(-(accr[v] + br)));
      float zz = 1.f / (1.f + __expf(-(accz[v] + bz)));
      float nn = tanhf(accin[v] + bin + r * (acchn[v] + bhn));
      float hold = hf32[nm * D_H + f];
      float hnew = (1.f - zz) * nn + zz * hold;
      hf32[nm * D_H + f]    = hnew;
      hbf_nxt[nm * D_H + f] = f2bf(hnew);
    }
  }
}

// ---------- output projection ----------
__global__ __launch_bounds__(256) void out_kernel(
    const bf16* __restrict__ hbf, const unsigned short* __restrict__ woutf,
    const float* __restrict__ b_out, float* __restrict__ out)
{
  const int w  = threadIdx.x >> 5;
  const int l  = threadIdx.x & 31;
  const int lo = l & 15, hi = (l >> 4) & 1;
  const int tile = blockIdx.x * 8 + w;
  if (tile >= N_NTILES) return;
  const size_t node = (size_t)tile * 16 + lo;

  v16bf a[4];
  #pragma unroll
  for (int kt = 0; kt < 4; ++kt)
    a[kt] = lda_bf(hbf + node * D_H + kt * 32 + hi * 8);
  v8f acc[4];
  #pragma unroll
  for (int nt = 0; nt < 4; ++nt) acc[nt] = vzero8();
  #pragma unroll
  for (int kt = 0; kt < 4; ++kt)
    #pragma unroll
    for (int nt = 0; nt < 4; ++nt)
      acc[nt] = wmma_bf16(a[kt], ldb(woutf, nt * 4 + kt, l), acc[nt]);
  #pragma unroll
  for (int nt = 0; nt < 4; ++nt) {
    float bo = b_out[nt * 16 + lo];
    #pragma unroll
    for (int v = 0; v < 8; ++v)
      out[((size_t)tile * 16 + hi * 8 + v) * D_OUT + nt * 16 + lo] = acc[nt][v] + bo;
  }
}

extern "C" void kernel_launch(void* const* d_in, const int* in_sizes, int n_in,
                              void* d_out, int out_size, void* d_ws, size_t ws_size,
                              hipStream_t stream) {
  const float* node_inputs = (const float*)d_in[0];
  const int*   src_ids = (const int*)d_in[1];
  const int*   dst_ids = (const int*)d_in[2];
  const float* W1 = (const float*)d_in[3];
  const float* b1 = (const float*)d_in[4];
  const float* W2 = (const float*)d_in[5];
  const float* b2 = (const float*)d_in[6];
  const float* W3 = (const float*)d_in[7];
  const float* b3 = (const float*)d_in[8];
  const float* W_ih  = (const float*)d_in[9];
  const float* W_hh  = (const float*)d_in[10];
  const float* b_ih  = (const float*)d_in[11];
  const float* b_hh  = (const float*)d_in[12];
  const float* W_out = (const float*)d_in[13];
  const float* b_out = (const float*)d_in[14];
  float* out = (float*)d_out;

  char* ws = (char*)d_ws;
  size_t off = 0;
  auto take = [&](size_t bytes) {
    char* p = ws + off;
    off = (off + bytes + 255) & ~(size_t)255;
    return p;
  };
  float* hf32 = (float*)take((size_t)N_NODES * D_H * 4);
  bf16*  hbfA = (bf16*)take((size_t)N_NODES * D_H * 2);
  bf16*  hbfB = (bf16*)take((size_t)N_NODES * D_H * 2);
  float* agg  = (float*)take((size_t)N_NODES * D_EDGE * 4);
  unsigned short* ewf   = (unsigned short*)take(92160);           // w1|w2|w3
  unsigned short* wihf  = (unsigned short*)take((size_t)24 * 6 * 512 * 2);
  unsigned short* whhf  = (unsigned short*)take((size_t)24 * 4 * 512 * 2);
  unsigned short* woutf = (unsigned short*)take((size_t)4 * 4 * 512 * 2);
  unsigned short* w1f = ewf;
  unsigned short* w2f = ewf + 24576;
  unsigned short* w3f = ewf + 33792;

  // pack weights into WMMA B fragments (bf16)
  pack_b_kernel<<<6 * 8, 32, 0, stream>>>(W1,    w1f,   8, 2 * D_H);
  pack_b_kernel<<<6 * 3, 32, 0, stream>>>(W2,    w2f,   3, MLP_H);
  pack_b_kernel<<<8 * 3, 32, 0, stream>>>(W3,    w3f,   3, MLP_H);
  pack_b_kernel<<<24 * 6, 32, 0, stream>>>(W_ih, wihf,  6, D_IN + D_EDGE);
  pack_b_kernel<<<24 * 4, 32, 0, stream>>>(W_hh, whhf,  4, D_H);
  pack_b_kernel<<<4 * 4, 32, 0, stream>>>(W_out, woutf, 4, D_H);

  // h0 = 0
  zero_kernel<<<2048, 256, 0, stream>>>((unsigned int*)hf32, (long)N_NODES * D_H);
  zero_kernel<<<2048, 256, 0, stream>>>((unsigned int*)hbfA, (long)N_NODES * D_H / 2);

  const int edge_blocks = (N_ETILES + 8 * T_EDGE - 1) / (8 * T_EDGE);   // 782
  const int gru_blocks  = (N_NTILES + 8 * T_GRU - 1) / (8 * T_GRU);     // 49
  const int out_blocks  = (N_NTILES + 7) / 8;                           // 391

  for (int it = 0; it < N_ITERS; ++it) {
    bf16* hcur = (it & 1) ? hbfB : hbfA;
    bf16* hnxt = (it & 1) ? hbfA : hbfB;
    zero_kernel<<<2048, 256, 0, stream>>>((unsigned int*)agg, (long)N_NODES * D_EDGE);
    edge_mlp_kernel<<<edge_blocks, 256, 0, stream>>>(
        hcur, src_ids, dst_ids, ewf, b1, b2, b3, agg);
    gru_kernel<<<dim3(gru_blocks, 8), 256, 0, stream>>>(
        node_inputs, agg, hcur, hf32, hnxt, wihf, whhf, b_ih, b_hh);
    out_kernel<<<out_blocks, 256, 0, stream>>>(
        hnxt, woutf, b_out, out + (size_t)it * N_NODES * D_OUT);
  }
}
